// LongTermAttention_25391846654038
// MI455X (gfx1250) — matrix operand order repaired and checked by reading, not compiled, not run on hardware
//
#include <hip/hip_runtime.h>
#include <hip/hip_bf16.h>

// ---------- types ----------
typedef __bf16 v8bf  __attribute__((ext_vector_type(8)));
typedef __bf16 v16bf __attribute__((ext_vector_type(16)));
typedef float  v8f   __attribute__((ext_vector_type(8)));

#define NB_DIM 1024   // attn_num_basis (K)
#define D_DIM  1024   // value dim (N of GEMM)
#define BM 128
#define BN 256
#define BK 32
#define NITER (NB_DIM / BK)
#define NTHREADS 512

static __device__ __forceinline__ __bf16 f2bf(float x) {
    unsigned u = __builtin_bit_cast(unsigned, x);
    u += 0x7FFFu + ((u >> 16) & 1u);          // round-to-nearest-even
    unsigned short h = (unsigned short)(u >> 16);
    return __builtin_bit_cast(__bf16, h);
}

// ---------- kernel 1: Bv [NB, D] f32  ->  BvT [D, NB] bf16 (K contiguous) ----------
__global__ __launch_bounds__(256) void bv_transpose_bf16(
    const float* __restrict__ Bv, __bf16* __restrict__ BvT)
{
    __shared__ __bf16 tile[32][33];
    const int tx = threadIdx.x;            // 0..31
    const int ty = threadIdx.y;            // 0..7
    const int nt = blockIdx.x * 32;        // D tile
    const int kt = blockIdx.y * 32;        // NB tile
    #pragma unroll
    for (int j = 0; j < 32; j += 8)
        tile[ty + j][tx] = f2bf(Bv[(size_t)(kt + ty + j) * D_DIM + nt + tx]);
    __syncthreads();
    #pragma unroll
    for (int j = 0; j < 32; j += 8)
        BvT[(size_t)(nt + ty + j) * NB_DIM + kt + tx] = tile[tx][ty + j];
}

// ---------- kernel 2: fused r-generation + bf16 WMMA GEMM (double-buffered) ----------
__global__ __launch_bounds__(NTHREADS) void longterm_attn_gemm(
    const float* __restrict__ theta,        // [N, 2]
    const float* __restrict__ basis_mu,     // [NB]
    const float* __restrict__ basis_sigma,  // [NB]
    const __bf16* __restrict__ BvT,         // [D, NB] bf16
    float* __restrict__ out)                // [N, D]
{
    __shared__ __align__(16) __bf16 Alds[2][BM * BK];   // r tiles   [m][k]
    __shared__ __align__(16) __bf16 Blds[2][BN * BK];   // BvT tiles [n][k]
    __shared__ float s_mu[BM];
    __shared__ float s_sig[BM];
    __shared__ float s_bmu[NB_DIM];
    __shared__ float s_bs2[NB_DIM];

    const int t     = threadIdx.x;
    const int mbase = blockIdx.x * BM;   // row block of N
    const int nbase = blockIdx.y * BN;   // col block of D

    // per-row canonical params: sigma_sq = -0.5/theta1 ; mu = theta0*sigma_sq
    if (t < BM) {
        float t0  = theta[(size_t)(mbase + t) * 2 + 0];
        float t1  = theta[(size_t)(mbase + t) * 2 + 1];
        float sig = -0.5f / t1;
        s_sig[t] = sig;
        s_mu[t]  = t0 * sig;
    }
    for (int i = t; i < NB_DIM; i += NTHREADS) {
        s_bmu[i] = basis_mu[i];
        float bs = basis_sigma[i];
        s_bs2[i] = bs * bs;
    }
    __syncthreads();

    const int lane = t & 31;
    const int wid  = t >> 5;              // 0..15
    const int wm   = (wid & 3)  * 32;     // wave M offset within block
    const int wn   = (wid >> 2) * 64;     // wave N offset within block

    // A-staging assignment: each thread computes 8 consecutive k for one m
    const int am = t >> 2;                // 0..127
    const int ak = (t & 3) * 8;           // 0,8,16,24
    const float mu_m  = s_mu[am];
    const float sig_m = s_sig[am];

    // B-staging assignment: two 16B chunks per thread
    const int bn0 = t >> 1;               // 0..255  (unused; staging uses n = idx >> 2)
    // generic form below uses idx = t + p*NTHREADS

    v8f acc[2][4];
    #pragma unroll
    for (int i = 0; i < 2; ++i)
        #pragma unroll
        for (int j = 0; j < 4; ++j)
            acc[i][j] = (v8f){0.f, 0.f, 0.f, 0.f, 0.f, 0.f, 0.f, 0.f};

    const int lhalf = (lane < 16) ? 0 : 8;   // K half selector (contiguous halves)
    (void)bn0;

    // ---- staging helper: fill buffer `buf` with tiles for k-offset k0 ----
    auto stage = [&](int k0, int buf) {
        // A: compute r[m][k0..k0+31] in fp32, pack bf16 to LDS
        v8bf rp;
        #pragma unroll
        for (int j = 0; j < 8; ++j) {
            int k      = k0 + ak + j;
            float s2   = s_bs2[k] + sig_m;
            float invs = rsqrtf(s2);
            float z    = (mu_m - s_bmu[k]) * invs;
            float r    = 0.3989422804014327f * __expf(-0.5f * z * z) * invs;
            rp[j] = f2bf(r);
        }
        *(v8bf*)&Alds[buf][am * BK + ak] = rp;
        // B: copy BvT rows (K contiguous) into LDS
        #pragma unroll
        for (int p = 0; p < 2; ++p) {
            int idx = t + p * NTHREADS;
            int n   = idx >> 2;            // 0..255
            int q   = (idx & 3) * 8;       // 0,8,16,24
            *(v8bf*)&Blds[buf][n * BK + q] =
                *(const v8bf*)&BvT[(size_t)(nbase + n) * NB_DIM + k0 + q];
        }
    };

    stage(0, 0);                           // prologue: fill buffer 0

    for (int it = 0; it < NITER; ++it) {
        const int buf = it & 1;
        __syncthreads();                   // buffer `buf` fully staged; buf^1 free

        // stage next tile into the other buffer (overlaps with WMMA below)
        if (it + 1 < NITER)
            stage((it + 1) * BK, buf ^ 1);

        // ---- load fragments from buffer `buf` (contiguous K halves) ----
        // A 16x32: lane<16 -> K 0-15 ; lane>=16 -> K 16-31 (contiguous)
        v16bf afrag[2];
        #pragma unroll
        for (int i = 0; i < 2; ++i) {
            const __bf16* base = &Alds[buf][(wm + i * 16 + (lane & 15)) * BK + lhalf * 2];
            v8bf lo = *(const v8bf*)(base);       // K = 2*lhalf .. 2*lhalf+7
            v8bf hi = *(const v8bf*)(base + 8);   // K = 2*lhalf+8 ..
            afrag[i] = __builtin_shufflevector(lo, hi,
                0, 1, 2, 3, 4, 5, 6, 7, 8, 9, 10, 11, 12, 13, 14, 15);
        }
        // B 32x16: lane<16 -> K 0-15 ; lane>=16 -> K 16-31 (contiguous)
        v16bf bfrag[4];
        #pragma unroll
        for (int j = 0; j < 4; ++j) {
            const __bf16* base = &Blds[buf][(wn + j * 16 + (lane & 15)) * BK + lhalf * 2];
            bfrag[j] = *(const v16bf*)base;
        }

        // ---- 8 WMMAs per wave per K-step ----
        #pragma unroll
        for (int i = 0; i < 2; ++i)
            #pragma unroll
            for (int j = 0; j < 4; ++j)
                acc[i][j] = __builtin_amdgcn_wmma_f32_16x16x32_bf16(
                    false, afrag[i], false, bfrag[j],
                    (short)0, acc[i][j], false, false);
    }

    // ---- epilogue: C/D layout -> global (rows = VGPR idx (+8 upper half-wave)) ----
    #pragma unroll
    for (int i = 0; i < 2; ++i) {
        #pragma unroll
        for (int j = 0; j < 4; ++j) {
            int col   = nbase + wn + j * 16 + (lane & 15);
            int rbase = mbase + wm + i * 16 + ((lane < 16) ? 0 : 8);
            #pragma unroll
            for (int v = 0; v < 8; ++v)
                out[(size_t)(rbase + v) * D_DIM + col] = acc[i][j][v];
        }
    }
}

extern "C" void kernel_launch(void* const* d_in, const int* in_sizes, int n_in,
                              void* d_out, int out_size, void* d_ws, size_t ws_size,
                              hipStream_t stream) {
    const float* theta       = (const float*)d_in[0];   // [N,2]
    const float* basis_mu    = (const float*)d_in[1];   // [1024]
    const float* basis_sigma = (const float*)d_in[2];   // [1024]
    const float* Bv          = (const float*)d_in[3];   // [1024,1024]
    float* out               = (float*)d_out;           // [N,1024]
    const int N = in_sizes[0] / 2;                      // 65536

    __bf16* BvT = (__bf16*)d_ws;                        // 2 MB bf16, L2-resident

    dim3 tb(32, 8);
    dim3 tg(D_DIM / 32, NB_DIM / 32);
    bv_transpose_bf16<<<tg, tb, 0, stream>>>(Bv, BvT);

    dim3 gb(N / BM, D_DIM / BN);                        // 512 x 4
    longterm_attn_gemm<<<gb, NTHREADS, 0, stream>>>(
        theta, basis_mu, basis_sigma, BvT, out);
}
